// Quantizer_31653908971537
// MI455X (gfx1250) — matrix operand order repaired and checked by reading, not compiled, not run on hardware
//
#include <hip/hip_runtime.h>
#include <hip/hip_bf16.h>

typedef __attribute__((ext_vector_type(16))) __bf16 v16bf;
typedef __attribute__((ext_vector_type(8)))  float  v8f;

#define D_DIM 512
#define NT 16               // codes per chunk (one 16x16 WMMA tile of scores)
#define PAD_ROWS (2 * NT)   // scratch padding rows so pipeline/prefetch needs no clamps
#define ROWS_PER_WAVE 16
#define WAVES_PER_BLOCK 8   // 256 threads = 8 wave32

union AFrag {
    v16bf v;
    unsigned short s[16];
    uint4 q[2];
};

static __device__ __forceinline__ unsigned short f2bf(float f) {
    unsigned int u = __float_as_uint(f);
    unsigned int r = (u + 0x7FFFu + ((u >> 16) & 1u)) >> 16;  // round-to-nearest-even
    return (unsigned short)r;
}

// ---------------------------------------------------------------------------
// Kernel 0: codebook f32 -> bf16 (scratch), per-code squared norms, zero loss.
// ---------------------------------------------------------------------------
__global__ void __launch_bounds__(256) vq_prep_kernel(
    const float* __restrict__ emb, unsigned short* __restrict__ ebf,
    float* __restrict__ enorm, float* __restrict__ loss_out) {
    __shared__ float red[256];
    const int k = blockIdx.x;
    const int t = threadIdx.x;
    const float* row = emb + (size_t)k * D_DIM;
    float a = row[2 * t];
    float b = row[2 * t + 1];
    unsigned int packed = (unsigned int)f2bf(a) | ((unsigned int)f2bf(b) << 16);
    ((unsigned int*)(ebf + (size_t)k * D_DIM))[t] = packed;
    red[t] = a * a + b * b;
    __syncthreads();
    for (int off = 128; off > 0; off >>= 1) {
        if (t < off) red[t] += red[t + off];
        __syncthreads();
    }
    if (t == 0) {
        enorm[k] = red[0];
        if (k == 0) loss_out[0] = 0.0f;   // reset loss accumulator every launch
    }
}

// ---------------------------------------------------------------------------
// Kernel 1: fused score GEMM + running argmax per row.
// score = z.e - 0.5*||e||^2  (argmax score == argmin distance).
// A-tile (16x512 bf16) in registers; B streamed through a 4-deep software
// pipeline; two WMMA chains starting from inline C=0; -0.5||e||^2 folded
// into the per-chunk argmax epilogue via fma.
// ---------------------------------------------------------------------------
__global__ void __launch_bounds__(256) vq_argmin_kernel(
    const float* __restrict__ z, const unsigned short* __restrict__ ebf,
    const float* __restrict__ enorm, int* __restrict__ idxbuf, int K) {
    const int lane = threadIdx.x & 31;
    const int wave = threadIdx.x >> 5;
    const int rowBase = blockIdx.x * (WAVES_PER_BLOCK * ROWS_PER_WAVE) + wave * ROWS_PER_WAVE;
    const int m    = lane & 15;
    const int hi   = lane >> 4;          // 0: lanes 0-15, 1: lanes 16-31
    const int koffA = hi ? 16 : 0;       // this code's A k order (contiguous): K 0..15 vs 16..31
    const int koffB = hi ? 16 : 0;       // this code's B k order (same as A): K 0..15 vs 16..31

    // Load this wave's A fragments (16 rows x 512 d) once, keep in registers.
    AFrag a[16];
    {
        const float* zrow = z + (size_t)(rowBase + m) * D_DIM;
#pragma unroll
        for (int j = 0; j < 16; ++j) {
            const int d0 = j * 32 + koffA;
            float4 p0 = *(const float4*)(zrow + d0);
            float4 p1 = *(const float4*)(zrow + d0 + 4);
            float4 p2 = *(const float4*)(zrow + d0 + 8);
            float4 p3 = *(const float4*)(zrow + d0 + 12);
            a[j].s[0]  = f2bf(p0.x); a[j].s[1]  = f2bf(p0.y);
            a[j].s[2]  = f2bf(p0.z); a[j].s[3]  = f2bf(p0.w);
            a[j].s[4]  = f2bf(p1.x); a[j].s[5]  = f2bf(p1.y);
            a[j].s[6]  = f2bf(p1.z); a[j].s[7]  = f2bf(p1.w);
            a[j].s[8]  = f2bf(p2.x); a[j].s[9]  = f2bf(p2.y);
            a[j].s[10] = f2bf(p2.z); a[j].s[11] = f2bf(p2.w);
            a[j].s[12] = f2bf(p3.x); a[j].s[13] = f2bf(p3.y);
            a[j].s[14] = f2bf(p3.z); a[j].s[15] = f2bf(p3.w);
        }
    }

    float best[8];
    int   bidx[8];
#pragma unroll
    for (int r = 0; r < 8; ++r) { best[r] = -3.4e38f; bidx[r] = 0; }

    // ---- B-fragment software pipeline, depth 4 ----
    const unsigned short* brow = ebf + (size_t)m * D_DIM;   // current chunk, this lane's column
    AFrag bbuf[4];
#pragma unroll
    for (int j = 0; j < 4; ++j) {
        const uint4* bp = (const uint4*)(brow + j * 32 + koffB);
        bbuf[j].q[0] = bp[0];
        bbuf[j].q[1] = bp[1];
    }

    const float* enp = enorm + m;
    float en_cur = enp[0];                // pipelined ||e||^2 for this lane's column

    for (int cb = 0; cb < K; cb += NT) {
        const int code = cb + m;
        const unsigned short* brow_next = brow + (size_t)NT * D_DIM;  // padded: always valid

        // Next chunk's norm (padded region is valid-but-unused on last iter).
        const float en_next = enp[cb + NT];
        // Prefetch two chunks ahead (also covered by padding).
        __builtin_prefetch(brow_next + (size_t)NT * D_DIM, 0, 3);

        v8f acc0 = {}, acc1 = {};         // both chains start at inline C=0

#pragma unroll
        for (int j = 0; j < 16; ++j) {                  // D = 512 = 16 x K32
            if (j & 1)
                acc1 = __builtin_amdgcn_wmma_f32_16x16x32_bf16(
                    false, a[j].v, false, bbuf[j & 3].v, (short)0, acc1, false, false);
            else
                acc0 = __builtin_amdgcn_wmma_f32_16x16x32_bf16(
                    false, a[j].v, false, bbuf[j & 3].v, (short)0, acc0, false, false);

            // Refill this slot for stage j+4 (statically: j<12 -> this chunk,
            // j>=12 -> next chunk's fragments 0..3), keeping 8 loads in flight.
            const unsigned short* src = (j + 4 < 16)
                ? (brow + (j + 4) * 32 + koffB)
                : (brow_next + (j + 4 - 16) * 32 + koffB);
            const uint4* bp = (const uint4*)src;
            bbuf[j & 3].q[0] = bp[0];
            bbuf[j & 3].q[1] = bp[1];
        }

#pragma unroll
        for (int r = 0; r < 8; ++r) {
            float s = fmaf(-0.5f, en_cur, acc0[r] + acc1[r]);
            if (s > best[r]) { best[r] = s; bidx[r] = code; }
        }
        en_cur = en_next;
        brow   = brow_next;
    }

    // Cross-lane arg-reduction within each half-wave (same output row set).
#pragma unroll
    for (int msk = 1; msk < 16; msk <<= 1) {
#pragma unroll
        for (int r = 0; r < 8; ++r) {
            float os = __shfl_xor(best[r], msk, 16);
            int   oi = __shfl_xor(bidx[r], msk, 16);
            if (os > best[r] || (os == best[r] && oi < bidx[r])) {
                best[r] = os; bidx[r] = oi;
            }
        }
    }
    if (m == 0) {
#pragma unroll
        for (int r = 0; r < 8; ++r)
            idxbuf[rowBase + hi * 8 + r] = bidx[r];     // lanes0-15: rows r; 16-31: r+8
    }
}

// ---------------------------------------------------------------------------
// Kernel 2: gather e[idx] (f32), write quantized_st = z + (q - z), indices,
// and accumulate loss = 1.25 * mean((q - z)^2).  One wave per row.
// ---------------------------------------------------------------------------
__global__ void __launch_bounds__(256) vq_gather_kernel(
    const float* __restrict__ z, const float* __restrict__ emb,
    const int* __restrict__ idxbuf, float* __restrict__ out, int N) {
    const int lane = threadIdx.x & 31;
    const int wave = threadIdx.x >> 5;
    const int row  = blockIdx.x * 8 + wave;
    const int idx  = idxbuf[row];

    const float4* z4 = (const float4*)(z   + (size_t)row * D_DIM);
    const float4* e4 = (const float4*)(emb + (size_t)idx * D_DIM);
    float4*       o4 = (float4*)(out + 1 + (size_t)row * D_DIM);

    float ss = 0.0f;
#pragma unroll
    for (int kk = 0; kk < 4; ++kk) {
        const int i = lane + kk * 32;
        float4 zv = z4[i];
        float4 ev = e4[i];
        float dx = ev.x - zv.x, dy = ev.y - zv.y;
        float dzz = ev.z - zv.z, dw = ev.w - zv.w;
        float4 st;                                   // z + sg(q - z), as in reference
        st.x = zv.x + dx; st.y = zv.y + dy; st.z = zv.z + dzz; st.w = zv.w + dw;
        o4[i] = st;
        ss += dx * dx + dy * dy + dzz * dzz + dw * dw;
    }
    for (int msk = 1; msk < 32; msk <<= 1) ss += __shfl_xor(ss, msk, 32);
    if (lane == 0) {
        atomicAdd(out, ss * (1.25f / ((float)N * (float)D_DIM)));
        out[1 + (size_t)N * D_DIM + row] = (float)idx;
    }
}

// ---------------------------------------------------------------------------
extern "C" void kernel_launch(void* const* d_in, const int* in_sizes, int n_in,
                              void* d_out, int out_size, void* d_ws, size_t ws_size,
                              hipStream_t stream) {
    (void)n_in; (void)out_size; (void)ws_size;
    const float* z   = (const float*)d_in[0];   // [N, 512]
    const float* emb = (const float*)d_in[1];   // [K, 512]
    const int N = in_sizes[0] / D_DIM;          // 32768
    const int K = in_sizes[1] / D_DIM;          // 4096

    // Scratch layout (codebook + norms padded by PAD_ROWS so the software
    // pipeline / prefetch never needs bounds checks):
    unsigned short* ebf = (unsigned short*)d_ws;                          // (K+PAD)*512 bf16
    size_t ebf_bytes   = (size_t)(K + PAD_ROWS) * D_DIM * 2;
    float* enorm = (float*)((char*)d_ws + ebf_bytes);                     // (K+PAD) floats
    size_t enorm_bytes = (size_t)(K + PAD_ROWS) * 4;
    int* idxbuf  = (int*)((char*)d_ws + ebf_bytes + enorm_bytes);         // N ints
    float* out   = (float*)d_out;  // [0]=loss, [1..]=quantized_st, tail=indices

    vq_prep_kernel<<<K, 256, 0, stream>>>(emb, ebf, enorm, out);
    vq_argmin_kernel<<<N / (WAVES_PER_BLOCK * ROWS_PER_WAVE), 256, 0, stream>>>(
        z, ebf, enorm, idxbuf, K);
    vq_gather_kernel<<<N / 8, 256, 0, stream>>>(z, emb, idxbuf, out, N);
}
